// NaViT_52149492908580
// MI455X (gfx1250) — compile-verified, matrix operand order checked by reading, not run on hardware
//
#include <hip/hip_runtime.h>
#include <cstdint>
#include <cstddef>

typedef __attribute__((ext_vector_type(16))) __bf16 v16bf;
typedef __attribute__((ext_vector_type(8)))  __bf16 v8bf;
typedef __attribute__((ext_vector_type(8)))  float  v8f;

constexpr int kB = 2, kN = 1024, kDim = 768, kHeads = 12, kDh = 64;
constexpr int kMlp = 3072, kPd = 768, kInner = kHeads * kDh, kDepth = 8;
constexpr int kM = kB * kN;   // 2048 token rows

enum { FLAG_BIAS = 1, FLAG_GELU = 2, FLAG_RES = 4 };

// ---------------------------------------------------------------------------
// 16x32 bf16 fragment loader (CDNA5 16-bit A layout; identical layout used
// for B operands stored contraction-contiguous).  Lane (m = lane&15,
// half = lane>>4) holds two contiguous 8-element runs: K = half*8 + [0..7]
// and K = 16 + half*8 + [0..7]  ->  two b128 loads (global or ds).
// ---------------------------------------------------------------------------
__device__ __forceinline__ v16bf load_frag16(const __bf16* base, int stride, int lane) {
  const int m = lane & 15, h = (lane >> 4) & 1;
  const __bf16* p = base + (size_t)m * stride + h * 8;
  v8bf lo = *reinterpret_cast<const v8bf*>(p);
  v8bf hi = *reinterpret_cast<const v8bf*>(p + 16);
  return __builtin_shufflevector(lo, hi, 0, 1, 2, 3, 4, 5, 6, 7,
                                 8, 9, 10, 11, 12, 13, 14, 15);
}

__device__ __forceinline__ v8f wmma_bf16(v16bf a, v16bf b, v8f c) {
  return __builtin_amdgcn_wmma_f32_16x16x32_bf16(false, a, false, b,
                                                 (short)0, c, false, false);
}

// ---------------------------------------------------------------------------
// 16-byte global -> LDS staging copy.  Prefer the CDNA5 async-DMA path
// (ASYNCcnt-tracked, no VGPR round trip); fall back to load+ds_store if the
// builtin is unavailable on this toolchain.  The builtin's parameters are
// pointers to int32x4 in AS1 (global) / AS3 (LDS), per the compiler's own
// diagnostic.
// ---------------------------------------------------------------------------
#if __has_builtin(__builtin_amdgcn_global_load_async_to_lds_b128)
#define GEMM_ASYNC_COPY 1
typedef int v4i32 __attribute__((vector_size(16)));
typedef __attribute__((address_space(1))) v4i32 global_v4i32;
typedef __attribute__((address_space(3))) v4i32 lds_v4i32;
#else
#define GEMM_ASYNC_COPY 0
#endif

__device__ __forceinline__ void copy16_g2l(const __bf16* gsrc, __bf16* ldst) {
#if GEMM_ASYNC_COPY
  __builtin_amdgcn_global_load_async_to_lds_b128(
      (global_v4i32*)(void*)gsrc,
      (lds_v4i32*)(void*)ldst,
      0, 0);
#else
  *reinterpret_cast<v8bf*>(ldst) = *reinterpret_cast<const v8bf*>(gsrc);
#endif
}

__device__ __forceinline__ void wait_async0() {
#if GEMM_ASYNC_COPY
#if __has_builtin(__builtin_amdgcn_s_wait_asynccnt)
  __builtin_amdgcn_s_wait_asynccnt(0);
#else
  asm volatile("s_wait_asynccnt 0" ::: "memory");
#endif
#endif
}

// ---------------------------------------------------------------------------
// fp32 [K,N] -> bf16 [N,K] tiled transpose+convert via LDS (coalesced both
// directions).  All weight dims are multiples of 32.  The bf16 weight arena
// (~114 MB) then lives in the 192 MB L2 across the whole forward pass.
// ---------------------------------------------------------------------------
__global__ __launch_bounds__(256)
void transpose_cvt_kernel(const float* __restrict__ src,
                          __bf16* __restrict__ dst, int K, int N) {
  __shared__ __bf16 tile[32][33];
  const int tx = threadIdx.x & 31, ty = threadIdx.x >> 5;   // 32x8 threads
  const int n0 = blockIdx.x * 32, k0 = blockIdx.y * 32;
#pragma unroll
  for (int i = 0; i < 32; i += 8)
    tile[ty + i][tx] = (__bf16)src[(size_t)(k0 + ty + i) * N + (n0 + tx)];
  __syncthreads();
#pragma unroll
  for (int i = 0; i < 32; i += 8)
    dst[(size_t)(n0 + ty + i) * K + (k0 + tx)] = tile[tx][ty + i];
}

// x = patches + pos_h[h_idx] + pos_w[w_idx]  -> bf16
__global__ void embed_kernel(const float* __restrict__ patches,
                             const int* __restrict__ h_idx,
                             const int* __restrict__ w_idx,
                             const float* __restrict__ pos_h,
                             const float* __restrict__ pos_w,
                             __bf16* __restrict__ out) {
  size_t i = (size_t)blockIdx.x * blockDim.x + threadIdx.x;
  size_t total = (size_t)kM * kPd;
  if (i >= total) return;
  int p = (int)(i % kPd);
  int t = (int)(i / kPd);
  float v = patches[i] + pos_h[(size_t)h_idx[t] * kPd + p]
                       + pos_w[(size_t)w_idx[t] * kPd + p];
  out[i] = (__bf16)v;
}

// ---------------------------------------------------------------------------
// WMMA GEMM: out = A[M,K](bf16) @ Wt[N,K](bf16, pre-transposed) (+bias)(gelu)
// (+res).  Block = 8 waves -> 256x64 tile; each wave owns 32x64 (2 A frags,
// 8 accumulators).
//   * The 64x32 B tile is shared by all 8 waves: staged once per k-step into
//     LDS (double-buffered) via async global->LDS copies (one b128 chunk per
//     thread), overlapped with the current step's WMMAs, then read back as
//     ds_load_b128 fragments.  Cuts per-block VMEM traffic ~2.4x.
//   * A fragments are register double-buffered from global (rows are
//     wave-private).
// ---------------------------------------------------------------------------
__global__ __launch_bounds__(256)
void gemm_wmma_kernel(const __bf16* __restrict__ A, const __bf16* __restrict__ Wt,
                      const float* __restrict__ bias, const float* __restrict__ res,
                      float* __restrict__ outF, __bf16* __restrict__ outB,
                      int M, int N, int K, int flags) {
  __shared__ __align__(16) __bf16 btile[2][64][40];   // 2 x 64 cols x 32 k (+pad)
  const int tid  = threadIdx.x;
  const int lane = tid & 31;
  const int wave = tid >> 5;
  const int row0 = blockIdx.y * 256 + wave * 32;
  const int col0 = blockIdx.x * 64;
  const int m = lane & 15, half = lane >> 4;
  (void)M;

  const __bf16* Ar0 = A + (size_t)row0 * K;
  const __bf16* Ar1 = A + (size_t)(row0 + 16) * K;

  // staging-copy assignment: thread copies 16B; 256 threads cover 64x32 bf16
  const int crow = tid >> 2, csub = (tid & 3) * 8;
  const __bf16* gsrc = Wt + (size_t)(col0 + crow) * K + csub;
  __bf16* ldst0 = &btile[0][crow][csub];
  __bf16* ldst1 = &btile[1][crow][csub];

  if (flags & FLAG_RES)
    __builtin_prefetch(res + (size_t)row0 * N + col0, 0, 1);

  v8f acc[2][4] = {};

  // prologue: stage k0=0 B tile, preload first A fragments
  copy16_g2l(gsrc, ldst0);
  v16bf a0 = load_frag16(Ar0, K, lane);
  v16bf a1 = load_frag16(Ar1, K, lane);
  wait_async0();
  __syncthreads();

  int cur = 0;
  for (int k0 = 0; k0 < K; k0 += 32) {
    const int kn = k0 + 32;
    // issue next B-tile staging copy + next A loads (overlap with WMMAs)
    if (kn < K) copy16_g2l(gsrc + kn, cur ? ldst0 : ldst1);
    v16bf na0 = a0, na1 = a1;
    if (kn < K) {
      na0 = load_frag16(Ar0 + kn, K, lane);
      na1 = load_frag16(Ar1 + kn, K, lane);
    }
    // B fragments from the staged LDS tile (ds_load_b128 pairs)
    const __bf16* lb = &btile[cur][0][0];
    v16bf b0 = load_frag16(lb + 0 * 16 * 40, 40, lane);
    v16bf b1 = load_frag16(lb + 1 * 16 * 40, 40, lane);
    v16bf b2 = load_frag16(lb + 2 * 16 * 40, 40, lane);
    v16bf b3 = load_frag16(lb + 3 * 16 * 40, 40, lane);

    acc[0][0] = wmma_bf16(a0, b0, acc[0][0]);
    acc[1][0] = wmma_bf16(a1, b0, acc[1][0]);
    acc[0][1] = wmma_bf16(a0, b1, acc[0][1]);
    acc[1][1] = wmma_bf16(a1, b1, acc[1][1]);
    acc[0][2] = wmma_bf16(a0, b2, acc[0][2]);
    acc[1][2] = wmma_bf16(a1, b2, acc[1][2]);
    acc[0][3] = wmma_bf16(a0, b3, acc[0][3]);
    acc[1][3] = wmma_bf16(a1, b3, acc[1][3]);

    a0 = na0; a1 = na1;
    wait_async0();       // next tile landed
    __syncthreads();     // all waves done reading btile[cur]
    cur ^= 1;
  }

#pragma unroll
  for (int rt = 0; rt < 2; ++rt) {
#pragma unroll
    for (int c = 0; c < 4; ++c) {
#pragma unroll
      for (int r = 0; r < 8; ++r) {
        int row = row0 + rt * 16 + r + 8 * half;   // C layout: M = r + 8*half
        int col = col0 + 16 * c + m;               //           N = lane&15
        float v = acc[rt][c][r];
        if (flags & FLAG_BIAS) v += bias[col];
        if (flags & FLAG_GELU) v = 0.5f * v * (1.0f + erff(v * 0.70710678118f));
        if (flags & FLAG_RES)  v += res[(size_t)row * N + col];
        if (outF) outF[(size_t)row * N + col] = v;
        if (outB) outB[(size_t)row * N + col] = (__bf16)v;
      }
    }
  }
}

// Row LayerNorm: fp32 in, optional bf16 and/or fp32 out.
__global__ __launch_bounds__(256)
void layernorm_kernel(const float* __restrict__ x, const float* __restrict__ g,
                      __bf16* __restrict__ outB, float* __restrict__ outF, int D) {
  __shared__ float s1[256], s2[256];
  const int row = blockIdx.x;
  const float* xr = x + (size_t)row * D;
  float sum = 0.f, sq = 0.f;
  for (int j = threadIdx.x; j < D; j += 256) { float v = xr[j]; sum += v; sq += v * v; }
  s1[threadIdx.x] = sum; s2[threadIdx.x] = sq;
  __syncthreads();
  for (int o = 128; o > 0; o >>= 1) {
    if ((int)threadIdx.x < o) { s1[threadIdx.x] += s1[threadIdx.x + o];
                                s2[threadIdx.x] += s2[threadIdx.x + o]; }
    __syncthreads();
  }
  float mean = s1[0] / D;
  float var  = s2[0] / D - mean * mean;
  float rstd = rsqrtf(var + 1e-5f);
  for (int j = threadIdx.x; j < D; j += 256) {
    float v = (xr[j] - mean) * rstd * g[j];
    if (outB) outB[(size_t)row * D + j] = (__bf16)v;
    if (outF) outF[(size_t)row * D + j] = v;
  }
}

// Per-(token,head) rms-normalize (F.normalize * sqrt(DH) * gamma) or plain
// convert; one wave handles one (token,head) (64 elems -> 2/lane).
// transpose_out=1 writes [B,H,DH,N] (V layout for contiguous B fragments).
__global__ __launch_bounds__(256)
void head_norm_kernel(const float* __restrict__ src, int row_stride,
                      const float* __restrict__ gamma, __bf16* __restrict__ dst,
                      int do_norm, int transpose_out) {
  const int lane = threadIdx.x & 31;
  const int w = (int)((blockIdx.x * blockDim.x + threadIdx.x) >> 5);
  const int h = w % kHeads;
  const int t = w / kHeads;
  const int b = t / kN, n = t % kN;
  const float* s = src + (size_t)t * row_stride + h * kDh;
  float v0 = s[lane], v1 = s[lane + 32];
  float scale = 1.0f;
  if (do_norm) {
    float ss = v0 * v0 + v1 * v1;
#pragma unroll
    for (int o = 16; o > 0; o >>= 1) ss += __shfl_xor(ss, o);
    scale = 8.0f / fmaxf(sqrtf(ss), 1e-12f);   // sqrt(DH)=8
  }
  float g0 = gamma ? gamma[h * kDh + lane]      : 1.0f;
  float g1 = gamma ? gamma[h * kDh + lane + 32] : 1.0f;
  size_t bh = (size_t)(b * kHeads + h);
  if (transpose_out) {
    dst[(bh * kDh + lane)      * kN + n] = (__bf16)(v0 * scale * g0);
    dst[(bh * kDh + lane + 32) * kN + n] = (__bf16)(v1 * scale * g1);
  } else {
    dst[(bh * kN + n) * kDh + lane]      = (__bf16)(v0 * scale * g0);
    dst[(bh * kN + n) * kDh + lane + 32] = (__bf16)(v1 * scale * g1);
  }
}

// ---------------------------------------------------------------------------
// Flash attention, one wave per 16-query tile. Block-diagonal image_ids mask,
// online softmax in the WMMA C layout, P bounced C->A through per-wave LDS
// fenced with s_wait_dscnt, P@V with 4 WMMAs into a 16x64 O accumulator.
// V fragments are loaded right after K fragments so they are in flight
// during the score WMMAs and the softmax VALU work.
// ---------------------------------------------------------------------------
__global__ __launch_bounds__(256)
void attn_kernel(const __bf16* __restrict__ qb, const __bf16* __restrict__ kb,
                 const __bf16* __restrict__ vt, const int* __restrict__ image_ids,
                 __bf16* __restrict__ ob) {
  __shared__ __align__(16) __bf16 pbuf[8][16][40];   // per-wave 16x32 (+pad)
  const int lane = threadIdx.x & 31;
  const int wave = threadIdx.x >> 5;
  const int gw = blockIdx.x * 8 + wave;
  const int qt = gw & 63;                  // N/16 = 64 query tiles
  const int bh = gw >> 6;                  // 0..B*H-1
  const int h = bh % kHeads, b = bh / kHeads;
  const int q0 = qt * 16;
  const int m = lane & 15, half = lane >> 4;

  const __bf16* Q  = qb + (size_t)bh * kN * kDh;     // [N,DH]
  const __bf16* Km = kb + (size_t)bh * kN * kDh;     // [N,DH]
  const __bf16* Vt = vt + (size_t)bh * kDh * kN;     // [DH,N]
  const int* ids = image_ids + b * kN;

  v16bf qf0 = load_frag16(Q + (size_t)q0 * kDh + 0,  kDh, lane);
  v16bf qf1 = load_frag16(Q + (size_t)q0 * kDh + 32, kDh, lane);

  int qid[8];
#pragma unroll
  for (int r = 0; r < 8; ++r) qid[r] = ids[q0 + r + 8 * half];

  float mrow[8], lrow[8];
#pragma unroll
  for (int r = 0; r < 8; ++r) { mrow[r] = -3.0e38f; lrow[r] = 0.0f; }
  v8f o0 = {}, o1 = {}, o2 = {}, o3 = {};

  for (int j0 = 0; j0 < kN; j0 += 32) {
    v16bf k00 = load_frag16(Km + (size_t)j0 * kDh + 0,  kDh, lane);
    v16bf k01 = load_frag16(Km + (size_t)j0 * kDh + 32, kDh, lane);
    v16bf k10 = load_frag16(Km + (size_t)(j0 + 16) * kDh + 0,  kDh, lane);
    v16bf k11 = load_frag16(Km + (size_t)(j0 + 16) * kDh + 32, kDh, lane);
    // V fragments: issue now, consume after softmax (latency overlap)
    v16bf vb0 = load_frag16(Vt + (size_t)0  * kN + j0, kN, lane);
    v16bf vb1 = load_frag16(Vt + (size_t)16 * kN + j0, kN, lane);
    v16bf vb2 = load_frag16(Vt + (size_t)32 * kN + j0, kN, lane);
    v16bf vb3 = load_frag16(Vt + (size_t)48 * kN + j0, kN, lane);

    v8f s0 = {}, s1 = {};
    s0 = wmma_bf16(qf0, k00, s0);
    s0 = wmma_bf16(qf1, k01, s0);
    s1 = wmma_bf16(qf0, k10, s1);
    s1 = wmma_bf16(qf1, k11, s1);

    const int kid0 = ids[j0 + m];
    const int kid1 = ids[j0 + 16 + m];
#pragma unroll
    for (int r = 0; r < 8; ++r) {
      float a0 = (qid[r] == kid0) ? s0[r] : -3.4e38f;
      float a1 = (qid[r] == kid1) ? s1[r] : -3.4e38f;
      float rmax = fmaxf(a0, a1);
#pragma unroll
      for (int o = 8; o > 0; o >>= 1) rmax = fmaxf(rmax, __shfl_xor(rmax, o));
      float mn = fmaxf(mrow[r], rmax);
      float p0 = __expf(a0 - mn);
      float p1 = __expf(a1 - mn);
      float rs = p0 + p1;
#pragma unroll
      for (int o = 8; o > 0; o >>= 1) rs += __shfl_xor(rs, o);
      float al = __expf(mrow[r] - mn);
      lrow[r] = lrow[r] * al + rs;
      mrow[r] = mn;
      pbuf[wave][r + 8 * half][m]      = (__bf16)p0;
      pbuf[wave][r + 8 * half][16 + m] = (__bf16)p1;
      o0[r] *= al; o1[r] *= al; o2[r] *= al; o3[r] *= al;
    }
    // DS stores above, DS reads below: fence the wave's LDS traffic.
    asm volatile("s_wait_dscnt 0\n\ts_wait_storecnt 0" ::: "memory");
    v16bf pf = load_frag16(&pbuf[wave][0][0], 40, lane);
    o0 = wmma_bf16(pf, vb0, o0);
    o1 = wmma_bf16(pf, vb1, o1);
    o2 = wmma_bf16(pf, vb2, o2);
    o3 = wmma_bf16(pf, vb3, o3);
  }

#pragma unroll
  for (int r = 0; r < 8; ++r) {
    float inv = 1.0f / fmaxf(lrow[r], 1e-30f);
    int q = q0 + r + 8 * half;
    size_t base = (size_t)(b * kN + q) * kInner + h * kDh;
    ob[base + 0  + m] = (__bf16)(o0[r] * inv);
    ob[base + 16 + m] = (__bf16)(o1[r] * inv);
    ob[base + 32 + m] = (__bf16)(o2[r] * inv);
    ob[base + 48 + m] = (__bf16)(o3[r] * inv);
  }
}

// ---------------------------------------------------------------------------
extern "C" void kernel_launch(void* const* d_in, const int* in_sizes, int n_in,
                              void* d_out, int out_size, void* d_ws, size_t ws_size,
                              hipStream_t stream) {
  (void)in_sizes; (void)n_in; (void)out_size; (void)ws_size;
  const float* patches  = (const float*)d_in[0];
  const int*   h_idx    = (const int*)d_in[1];
  const int*   w_idx    = (const int*)d_in[2];
  const int*   image_ids= (const int*)d_in[3];
  const float* pos_h    = (const float*)d_in[4];
  const float* pos_w    = (const float*)d_in[5];
  const float* Wp       = (const float*)d_in[6];
  const float* bp       = (const float*)d_in[7];
  const float* ln_p     = (const float*)d_in[8];
  const float* ln1      = (const float*)d_in[9];
  const float* qg       = (const float*)d_in[10];
  const float* kg       = (const float*)d_in[11];
  const float* Wq       = (const float*)d_in[12];
  const float* Wkv      = (const float*)d_in[13];
  const float* Wo       = (const float*)d_in[14];
  const float* ln2      = (const float*)d_in[15];
  const float* W1       = (const float*)d_in[16];
  const float* b1       = (const float*)d_in[17];
  const float* W2       = (const float*)d_in[18];
  const float* b2       = (const float*)d_in[19];
  const float* lnf      = (const float*)d_in[20];

  size_t off = 0;
  auto carve = [&](size_t elems, size_t eb) -> void* {
    off = (off + 255) & ~(size_t)255;
    void* p = (char*)d_ws + off;
    off += elems * eb;
    return p;
  };
  __bf16* WpT  = (__bf16*)carve((size_t)kDim * kPd, 2);
  __bf16* WqT  = (__bf16*)carve((size_t)kDepth * kInner * kDim, 2);
  __bf16* WkvT = (__bf16*)carve((size_t)kDepth * 2 * kInner * kDim, 2);
  __bf16* WoT  = (__bf16*)carve((size_t)kDepth * kDim * kInner, 2);
  __bf16* W1T  = (__bf16*)carve((size_t)kDepth * kMlp * kDim, 2);
  __bf16* W2T  = (__bf16*)carve((size_t)kDepth * kDim * kMlp, 2);
  __bf16* xemb = (__bf16*)carve((size_t)kM * kPd, 2);
  float*  t0   = (float*) carve((size_t)kM * kDim, 4);
  float*  x    = (float*) carve((size_t)kM * kDim, 4);
  __bf16* xn   = (__bf16*)carve((size_t)kM * kDim, 2);
  float*  qf   = (float*) carve((size_t)kM * kInner, 4);
  float*  kvf  = (float*) carve((size_t)kM * 2 * kInner, 4);
  __bf16* qb   = (__bf16*)carve((size_t)kB * kHeads * kN * kDh, 2);
  __bf16* kbb  = (__bf16*)carve((size_t)kB * kHeads * kN * kDh, 2);
  __bf16* vtb  = (__bf16*)carve((size_t)kB * kHeads * kDh * kN, 2);
  __bf16* obuf = (__bf16*)carve((size_t)kM * kInner, 2);
  __bf16* hbuf = (__bf16*)carve((size_t)kM * kMlp, 2);

  auto cvtT = [&](const float* src, __bf16* dst, int K, int N) {
    dim3 grid(N / 32, K / 32);
    transpose_cvt_kernel<<<grid, 256, 0, stream>>>(src, dst, K, N);
  };
  cvtT(Wp, WpT, kPd, kDim);
  for (int i = 0; i < kDepth; ++i) {
    cvtT(Wq  + (size_t)i * kDim * kInner,     WqT  + (size_t)i * kInner * kDim,     kDim, kInner);
    cvtT(Wkv + (size_t)i * kDim * 2 * kInner, WkvT + (size_t)i * 2 * kInner * kDim, kDim, 2 * kInner);
    cvtT(Wo  + (size_t)i * kInner * kDim,     WoT  + (size_t)i * kDim * kInner,     kInner, kDim);
    cvtT(W1  + (size_t)i * kDim * kMlp,       W1T  + (size_t)i * kMlp * kDim,       kDim, kMlp);
    cvtT(W2  + (size_t)i * kMlp * kDim,       W2T  + (size_t)i * kDim * kMlp,       kMlp, kDim);
  }

  {
    size_t total = (size_t)kM * kPd;
    embed_kernel<<<(int)((total + 255) / 256), 256, 0, stream>>>(
        patches, h_idx, w_idx, pos_h, pos_w, xemb);
  }

  auto gemm = [&](const __bf16* A, const __bf16* Wt, const float* bias,
                  const float* res, float* oF, __bf16* oB,
                  int M, int N, int K, int flags) {
    dim3 grid(N / 64, M / 256);
    gemm_wmma_kernel<<<grid, 256, 0, stream>>>(A, Wt, bias, res, oF, oB, M, N, K, flags);
  };

  gemm(xemb, WpT, bp, nullptr, t0, nullptr, kM, kDim, kPd, FLAG_BIAS);
  layernorm_kernel<<<kM, 256, 0, stream>>>(t0, ln_p, nullptr, x, kDim);

  const int hn_blocks = kB * kN * kHeads * 32 / 256;        // 3072
  const int attn_blocks = kB * kHeads * (kN / 16) / 8;      // 192

  for (int i = 0; i < kDepth; ++i) {
    layernorm_kernel<<<kM, 256, 0, stream>>>(x, ln1 + (size_t)i * kDim, xn, nullptr, kDim);
    gemm(xn, WqT  + (size_t)i * kInner * kDim,     nullptr, nullptr, qf,  nullptr, kM, kInner,     kDim, 0);
    gemm(xn, WkvT + (size_t)i * 2 * kInner * kDim, nullptr, nullptr, kvf, nullptr, kM, 2 * kInner, kDim, 0);
    head_norm_kernel<<<hn_blocks, 256, 0, stream>>>(qf,  kInner,     qg + (size_t)i * kHeads * kDh, qb,  1, 0);
    head_norm_kernel<<<hn_blocks, 256, 0, stream>>>(kvf, 2 * kInner, kg + (size_t)i * kHeads * kDh, kbb, 1, 0);
    head_norm_kernel<<<hn_blocks, 256, 0, stream>>>(kvf + kInner, 2 * kInner, nullptr, vtb, 0, 1);
    attn_kernel<<<attn_blocks, 256, 0, stream>>>(qb, kbb, vtb, image_ids, obuf);
    gemm(obuf, WoT + (size_t)i * kDim * kInner, nullptr, x, x, nullptr, kM, kDim, kInner, FLAG_RES);
    layernorm_kernel<<<kM, 256, 0, stream>>>(x, ln2 + (size_t)i * kDim, xn, nullptr, kDim);
    gemm(xn,   W1T + (size_t)i * kMlp * kDim, b1 + (size_t)i * kMlp, nullptr, nullptr, hbuf,
         kM, kMlp, kDim, FLAG_BIAS | FLAG_GELU);
    gemm(hbuf, W2T + (size_t)i * kDim * kMlp, b2 + (size_t)i * kDim, x, x, nullptr,
         kM, kDim, kMlp, FLAG_BIAS | FLAG_RES);
  }
  layernorm_kernel<<<kM, 256, 0, stream>>>(x, lnf, nullptr, (float*)d_out, kDim);
}
